// GARModelCell_14482629722612
// MI455X (gfx1250) — matrix operand order repaired and checked by reading, not compiled, not run on hardware
//
#include <hip/hip_runtime.h>
#include <math.h>

typedef __bf16 bf16;
typedef __attribute__((ext_vector_type(16))) __bf16 v16bf;
typedef __attribute__((ext_vector_type(8)))  float  v8f;

#define F_EMB 256
#define NHEAD 8
#define DH    32
#define NB    16
#define TT    1024
#define DFFD  1024
#define KEYT_C 256
#define SPAD  288      // 257 padded to multiple of 32
#define SVALID 257

// ---------------- basic conversions ----------------
__device__ __forceinline__ bf16 f2bf(float f){
  union { float f; unsigned u; } x; x.f = f;
  unsigned r = (x.u + 0x7FFFu + ((x.u >> 16) & 1u)) >> 16;
  unsigned short h = (unsigned short)r;
  bf16 b; __builtin_memcpy(&b, &h, 2); return b;
}

__device__ __forceinline__ v8f wmma_bf(v16bf a, v16bf b, v8f c){
  return __builtin_amdgcn_wmma_f32_16x16x32_bf16(false, a, false, b, (short)0, c, false, false);
}

// ---------------- WMMA fragment loaders (contiguous K order, same for A and B) ----------------
// A: 16x32 bf16, row-major source, contiguous K layout (k = 16*half + s)
__device__ __forceinline__ v16bf load_a_frag(const bf16* A, size_t row0, int k0, int lda, int lane){
  int half = lane >> 4, lid = lane & 15;
  const bf16* rp = A + (row0 + (size_t)lid) * (size_t)lda + k0;
  v16bf a;
  #pragma unroll
  for (int v = 0; v < 8; ++v){
    int kb = half*16 + 2*v;
    a[2*v] = rp[kb]; a[2*v+1] = rp[kb+1];
  }
  return a;
}

// B[k,n] = W[n,k], W row-major (C,K): weight-transpose fragment
__device__ __forceinline__ v16bf load_b_fragT(const bf16* W, int k0, size_t col0, int ldw, int lane){
  int half = lane >> 4, lid = lane & 15;
  const bf16* cp = W + (col0 + (size_t)lid) * (size_t)ldw + k0 + half*16;
  v16bf b;
  #pragma unroll
  for (int v = 0; v < 8; ++v){ b[2*v] = cp[2*v]; b[2*v+1] = cp[2*v+1]; }
  return b;
}

// B row-major (K,N): direct fragment (used for P@V with V rows = keys)
__device__ __forceinline__ v16bf load_b_fragN(const bf16* B, int col0, int ldb, int lane){
  int half = lane >> 4, lid = lane & 15;
  v16bf b;
  #pragma unroll
  for (int v = 0; v < 8; ++v){
    int k = half*16 + 2*v;
    b[2*v]   = B[(size_t)k * ldb + col0 + lid];
    b[2*v+1] = B[(size_t)(k+1) * ldb + col0 + lid];
  }
  return b;
}

// ---------------- elementwise kernels ----------------
__global__ void cvt_kernel(const float* __restrict__ s, bf16* __restrict__ d, int n){
  int i = blockIdx.x*blockDim.x + threadIdx.x;
  if (i < n) d[i] = f2bf(s[i]);
}

__global__ void zero_kernel(float* __restrict__ p, int n){
  int i = blockIdx.x*blockDim.x + threadIdx.x;
  if (i < n) p[i] = 0.f;
}

// (T,N,F) -> (N,T,F), fp32 + bf16
__global__ void transpose_kernel(const float* __restrict__ in, float* __restrict__ xF, bf16* __restrict__ xB){
  int i = blockIdx.x*blockDim.x + threadIdx.x;
  if (i >= TT*NB*F_EMB) return;
  int f = i & 255, n = (i >> 8) & 15, t = i >> 12;
  float v = in[i];
  size_t o = ((size_t)n*TT + t)*F_EMB + f;
  xF[o] = v; xB[o] = f2bf(v);
}

// one wave per row, F=256
__global__ void ln_kernel(const float* __restrict__ X, const float* __restrict__ g, const float* __restrict__ b,
                          float* __restrict__ outF, bf16* __restrict__ outB, int R){
  int wid  = (int)((blockIdx.x*blockDim.x + threadIdx.x) >> 5);
  int lane = threadIdx.x & 31;
  if (wid >= R) return;
  const float* row = X + (size_t)wid * F_EMB;
  float v[8]; float s = 0.f;
  #pragma unroll
  for (int i = 0; i < 8; ++i){ v[i] = row[lane + 32*i]; s += v[i]; }
  #pragma unroll
  for (int m = 16; m >= 1; m >>= 1) s += __shfl_xor(s, m, 32);
  float mean = s * (1.f/256.f);
  float var = 0.f;
  #pragma unroll
  for (int i = 0; i < 8; ++i){ float d = v[i] - mean; var += d*d; }
  #pragma unroll
  for (int m = 16; m >= 1; m >>= 1) var += __shfl_xor(var, m, 32);
  float inv = rsqrtf(var * (1.f/256.f) + 1e-5f);
  #pragma unroll
  for (int i = 0; i < 8; ++i){
    int c = lane + 32*i;
    float y = (v[i]-mean)*inv*g[c] + b[c];
    if (outF) outF[(size_t)wid*F_EMB + c] = y;
    if (outB) outB[(size_t)wid*F_EMB + c] = f2bf(y);
  }
}

// ---------------- WMMA GEMM: Y(R,C) = A(R,K) @ W(C,K)^T + bias [+resid] [relu] ----------------
// Each wave computes a 16x64 output strip: one A fragment reused across 4 B fragments.
// Software-pipelined: fragments for k-step i+1 are fetched before the WMMAs of step i wait.
__global__ void gemm_kernel(const bf16* __restrict__ A, const bf16* __restrict__ W,
                            const float* __restrict__ bias, const float* __restrict__ resid,
                            float* __restrict__ outF, bf16* __restrict__ outB,
                            int R, int K, int C, int relu){
  int lane = threadIdx.x & 31;
  int wave = (int)((blockIdx.x*blockDim.x + threadIdx.x) >> 5);
  int tn = C >> 6;                      // 64-col strips
  int nt = (R >> 4) * tn;
  if (wave >= nt) return;
  int tmi = wave / tn, tci = wave % tn;
  size_t row0 = (size_t)tmi * 16;
  int col0 = tci * 64;
  v8f acc0 = {}, acc1 = {}, acc2 = {}, acc3 = {};

  v16bf a  = load_a_frag(A, row0, 0, K, lane);
  v16bf b0 = load_b_fragT(W, 0, (size_t)col0,      K, lane);
  v16bf b1 = load_b_fragT(W, 0, (size_t)col0 + 16, K, lane);
  v16bf b2 = load_b_fragT(W, 0, (size_t)col0 + 32, K, lane);
  v16bf b3 = load_b_fragT(W, 0, (size_t)col0 + 48, K, lane);

  for (int k0 = 32; k0 < K; k0 += 32){
    // prefetch next k-step fragments (independent of current WMMAs)
    v16bf an  = load_a_frag(A, row0, k0, K, lane);
    v16bf bn0 = load_b_fragT(W, k0, (size_t)col0,      K, lane);
    v16bf bn1 = load_b_fragT(W, k0, (size_t)col0 + 16, K, lane);
    v16bf bn2 = load_b_fragT(W, k0, (size_t)col0 + 32, K, lane);
    v16bf bn3 = load_b_fragT(W, k0, (size_t)col0 + 48, K, lane);
    acc0 = wmma_bf(a, b0, acc0);
    acc1 = wmma_bf(a, b1, acc1);
    acc2 = wmma_bf(a, b2, acc2);
    acc3 = wmma_bf(a, b3, acc3);
    a = an; b0 = bn0; b1 = bn1; b2 = bn2; b3 = bn3;
  }
  acc0 = wmma_bf(a, b0, acc0);
  acc1 = wmma_bf(a, b1, acc1);
  acc2 = wmma_bf(a, b2, acc2);
  acc3 = wmma_bf(a, b3, acc3);

  int half = lane >> 4, lid = lane & 15;
  v8f accs[4] = { acc0, acc1, acc2, acc3 };
  #pragma unroll
  for (int j = 0; j < 4; ++j){
    #pragma unroll
    for (int r = 0; r < 8; ++r){
      size_t row = row0 + half*8 + r;
      int col = col0 + j*16 + lid;
      float v = accs[j][r] + bias[col];
      if (resid) v += resid[row*(size_t)C + col];
      if (relu)  v = fmaxf(v, 0.f);
      size_t oi = row*(size_t)C + col;
      if (outF) outF[oi] = v;
      if (outB) outB[oi] = f2bf(v);
    }
  }
}

// ---------------- flash attention: qkv (NB*S,768) bf16 -> o (NB*S,256) bf16 ----------------
__global__ void flash_attn_kernel(const bf16* __restrict__ qkv, bf16* __restrict__ o,
                                  int S, int Sv, int qtiles){
  __shared__ bf16 Pl[8][16*32];
  int wib  = threadIdx.x >> 5;
  int lane = threadIdx.x & 31;
  int wave = blockIdx.x * 8 + wib;
  int total = NB * NHEAD * qtiles;
  if (wave >= total) return;
  int n  = wave / (NHEAD*qtiles);
  int rem = wave % (NHEAD*qtiles);
  int h  = rem / qtiles;
  int qt = rem % qtiles;
  int half = lane >> 4, lid = lane & 15;
  const float scale = 0.17677669529663687f; // 1/sqrt(32)
  size_t rowbase = (size_t)n * S;

  v16bf qf = load_a_frag(qkv + h*DH, rowbase + (size_t)qt*16, 0, 768, lane);
  float m_run[8], l_run[8];
  v8f o0 = {}, o1 = {};
  #pragma unroll
  for (int r = 0; r < 8; ++r){ m_run[r] = -3.0e38f; l_run[r] = 0.f; }

  for (int kb = 0; kb < S/32; ++kb){
    int key0 = kb*32;
    v16bf kf0 = load_b_fragT(qkv + 256 + h*DH, 0, rowbase + key0,      768, lane);
    v16bf kf1 = load_b_fragT(qkv + 256 + h*DH, 0, rowbase + key0 + 16, 768, lane);
    v8f s0 = {}; s0 = wmma_bf(qf, kf0, s0);
    v8f s1 = {}; s1 = wmma_bf(qf, kf1, s1);
    bool msk0 = (key0 + lid)      >= Sv;
    bool msk1 = (key0 + 16 + lid) >= Sv;
    #pragma unroll
    for (int r = 0; r < 8; ++r){
      float a = msk0 ? -1e30f : s0[r]*scale;
      float b = msk1 ? -1e30f : s1[r]*scale;
      float mt = fmaxf(a, b);
      #pragma unroll
      for (int m = 8; m >= 1; m >>= 1) mt = fmaxf(mt, __shfl_xor(mt, m, 32));
      float mn = fmaxf(m_run[r], mt);
      float corr = __expf(m_run[r] - mn);
      float e0 = __expf(a - mn), e1 = __expf(b - mn);
      float rs = e0 + e1;
      #pragma unroll
      for (int m = 8; m >= 1; m >>= 1) rs += __shfl_xor(rs, m, 32);
      l_run[r] = l_run[r]*corr + rs;
      m_run[r] = mn;
      o0[r] *= corr; o1[r] *= corr;
      Pl[wib][(half*8 + r)*32 + lid]      = f2bf(e0);
      Pl[wib][(half*8 + r)*32 + 16 + lid] = f2bf(e1);
    }
    // P fragment from LDS (A layout, 16x32)
    v16bf pf;
    {
      const bf16* rp = &Pl[wib][lid*32];
      #pragma unroll
      for (int v = 0; v < 8; ++v){
        int kbx = half*16 + 2*v;
        pf[2*v] = rp[kbx]; pf[2*v+1] = rp[kbx+1];
      }
    }
    const bf16* Vb = qkv + (rowbase + key0)*768 + 512 + h*DH;
    v16bf vf0 = load_b_fragN(Vb, 0,  768, lane);
    v16bf vf1 = load_b_fragN(Vb, 16, 768, lane);
    o0 = wmma_bf(pf, vf0, o0);
    o1 = wmma_bf(pf, vf1, o1);
  }
  #pragma unroll
  for (int r = 0; r < 8; ++r){
    float inv = 1.f / l_run[r];
    size_t row = rowbase + (size_t)qt*16 + half*8 + r;
    o[row*F_EMB + h*DH + lid]      = f2bf(o0[r]*inv);
    o[row*F_EMB + h*DH + 16 + lid] = f2bf(o1[r]*inv);
  }
}

// ---------------- score MHA pass A: per-q-row softmax stats ----------------
__global__ void attn_stats_kernel(const bf16* __restrict__ qk, float* __restrict__ mO, float* __restrict__ lO,
                                  int qtiles){
  int lane = threadIdx.x & 31;
  int wave = (int)((blockIdx.x*blockDim.x + threadIdx.x) >> 5);
  if (wave >= NB*NHEAD*qtiles) return;
  int n = wave / (NHEAD*qtiles);
  int rem = wave % (NHEAD*qtiles);
  int h = rem / qtiles, qt = rem % qtiles;
  int half = lane >> 4;
  const float scale = 0.17677669529663687f;
  size_t rowbase = (size_t)n * TT;
  v16bf qf = load_a_frag(qk + h*DH, rowbase + (size_t)qt*16, 0, 512, lane);
  float m_run[8], l_run[8];
  #pragma unroll
  for (int r = 0; r < 8; ++r){ m_run[r] = -3.0e38f; l_run[r] = 0.f; }
  for (int kb = 0; kb < TT/32; ++kb){
    v16bf kf0 = load_b_fragT(qk + 256 + h*DH, 0, rowbase + kb*32,      512, lane);
    v16bf kf1 = load_b_fragT(qk + 256 + h*DH, 0, rowbase + kb*32 + 16, 512, lane);
    v8f s0 = {}; s0 = wmma_bf(qf, kf0, s0);
    v8f s1 = {}; s1 = wmma_bf(qf, kf1, s1);
    #pragma unroll
    for (int r = 0; r < 8; ++r){
      float a = s0[r]*scale, b = s1[r]*scale;
      float mt = fmaxf(a, b);
      #pragma unroll
      for (int m = 8; m >= 1; m >>= 1) mt = fmaxf(mt, __shfl_xor(mt, m, 32));
      float mn = fmaxf(m_run[r], mt);
      float rs = __expf(a - mn) + __expf(b - mn);
      #pragma unroll
      for (int m = 8; m >= 1; m >>= 1) rs += __shfl_xor(rs, m, 32);
      l_run[r] = l_run[r]*__expf(m_run[r] - mn) + rs;
      m_run[r] = mn;
    }
  }
  if ((lane & 15) == 0){
    #pragma unroll
    for (int r = 0; r < 8; ++r){
      int row = qt*16 + half*8 + r;
      mO[((size_t)n*NHEAD + h)*TT + row] = m_run[r];
      lO[((size_t)n*NHEAD + h)*TT + row] = l_run[r];
    }
  }
}

// ---------------- score MHA pass B: key importance = mean_h sum_q softmax ----------------
__global__ void attn_keyscore_kernel(const bf16* __restrict__ qk, const float* __restrict__ mI,
                                     const float* __restrict__ lI, float* __restrict__ kscore, int qtiles){
  int lane = threadIdx.x & 31;
  int wave = (int)((blockIdx.x*blockDim.x + threadIdx.x) >> 5);
  if (wave >= NB*NHEAD*qtiles) return;
  int n = wave / (NHEAD*qtiles);
  int rem = wave % (NHEAD*qtiles);
  int h = rem / qtiles, qt = rem % qtiles;
  int half = lane >> 4, lid = lane & 15;
  const float scale = 0.17677669529663687f;
  size_t rowbase = (size_t)n * TT;
  v16bf qf = load_a_frag(qk + h*DH, rowbase + (size_t)qt*16, 0, 512, lane);
  float mr[8], il[8];
  #pragma unroll
  for (int r = 0; r < 8; ++r){
    int row = qt*16 + half*8 + r;
    mr[r] = mI[((size_t)n*NHEAD + h)*TT + row];
    il[r] = 1.f / lI[((size_t)n*NHEAD + h)*TT + row];
  }
  for (int kb = 0; kb < TT/32; ++kb){
    v16bf kf0 = load_b_fragT(qk + 256 + h*DH, 0, rowbase + kb*32,      512, lane);
    v16bf kf1 = load_b_fragT(qk + 256 + h*DH, 0, rowbase + kb*32 + 16, 512, lane);
    v8f s0 = {}; s0 = wmma_bf(qf, kf0, s0);
    v8f s1 = {}; s1 = wmma_bf(qf, kf1, s1);
    float cs0 = 0.f, cs1 = 0.f;
    #pragma unroll
    for (int r = 0; r < 8; ++r){
      cs0 += __expf(s0[r]*scale - mr[r]) * il[r];
      cs1 += __expf(s1[r]*scale - mr[r]) * il[r];
    }
    cs0 += __shfl_xor(cs0, 16, 32);
    cs1 += __shfl_xor(cs1, 16, 32);
    if (half == 0){
      atomicAdd(&kscore[n*TT + kb*32 + lid],      cs0 * (1.f/NHEAD));
      atomicAdd(&kscore[n*TT + kb*32 + 16 + lid], cs1 * (1.f/NHEAD));
    }
  }
}

// ---------------- top-256 of 1024, output indices sorted ascending ----------------
__global__ void topk_kernel(const float* __restrict__ kscore, int* __restrict__ idxOut){
  int n = blockIdx.x, t = threadIdx.x;
  __shared__ float sv[TT];
  __shared__ int   fl[TT];
  __shared__ float rv[256];
  __shared__ int   ri[256];
  for (int i = t; i < TT; i += 256){ sv[i] = kscore[n*TT + i]; fl[i] = 0; }
  __syncthreads();
  for (int it = 0; it < KEYT_C; ++it){
    float bv = -3.4e38f; int bi = 0x7FFFFFFF;
    for (int i = t; i < TT; i += 256){ float v = sv[i]; if (v > bv){ bv = v; bi = i; } }
    rv[t] = bv; ri[t] = bi; __syncthreads();
    for (int s = 128; s > 0; s >>= 1){
      if (t < s){
        if (rv[t+s] > rv[t] || (rv[t+s] == rv[t] && ri[t+s] < ri[t])){ rv[t] = rv[t+s]; ri[t] = ri[t+s]; }
      }
      __syncthreads();
    }
    if (t == 0){ fl[ri[0]] = 1; sv[ri[0]] = -3.4e38f; }
    __syncthreads();
  }
  if (t == 0){
    int c = 0;
    for (int i = 0; i < TT; ++i) if (fl[i]) idxOut[n*KEYT_C + (c++)] = i;
  }
}

// ---------------- gather + PE + cls -> tm (NB,SPAD,F), zero padding ----------------
__global__ void build_tm_kernel(const float* __restrict__ x, const float* __restrict__ cls,
                                const int* __restrict__ idx, float* __restrict__ tm){
  int tid = blockIdx.x*blockDim.x + threadIdx.x;
  if (tid >= NB*SPAD*F_EMB) return;
  int f = tid & 255;
  int row = (tid >> 8) % SPAD;
  int n = tid / (SPAD*F_EMB);
  float v;
  if (row == 0) v = cls[f];
  else if (row <= KEYT_C){
    int j = row - 1;
    int src = idx[n*KEYT_C + j];
    int k = f >> 1;
    float div = __expf(-logf(10000.f) * (2.f*(float)k) / (float)F_EMB);
    float ang = (float)j * div;
    float pe = (f & 1) ? __cosf(ang) : __sinf(ang);
    v = x[((size_t)n*TT + src)*F_EMB + f] + pe;
  } else v = 0.f;
  tm[((size_t)n*SPAD + row)*F_EMB + f] = v;
}

__global__ void sfmean_kernel(const float* __restrict__ sa, const int* __restrict__ idx, float* __restrict__ out){
  int t = blockIdx.x*blockDim.x + threadIdx.x;
  if (t >= NB*F_EMB) return;
  int n = t >> 8, f = t & 255;
  float s = 0.f;
  for (int j = 0; j < KEYT_C; ++j){
    int src = idx[n*KEYT_C + j];
    s += sa[((size_t)n*TT + src)*F_EMB + f];
  }
  out[t] = s * (1.f/KEYT_C);
}

__global__ void finalize_kernel(const float* __restrict__ enc, const float* __restrict__ sfm, float* __restrict__ out){
  int t = blockIdx.x*blockDim.x + threadIdx.x;
  if (t >= NB*F_EMB) return;
  int n = t >> 8, f = t & 255;
  out[t] = enc[((size_t)n*SPAD)*F_EMB + f] + sfm[t];
}

// ---------------- host orchestration ----------------
struct EncParams {
  const bf16 *Wqkv, *Wo, *W1, *W2;
  const float *bqkv, *bo, *b1, *b2;
  const float *ln1g, *ln1b, *ln2g, *ln2b, *lnfg, *lnfb;
};

static void gemm_launch(hipStream_t s, const bf16* A, const bf16* W, const float* bias,
                        const float* resid, float* outF, bf16* outB, int R, int K, int C, int relu){
  int waves = (R/16)*(C/64);
  gemm_kernel<<<(waves*32 + 255)/256, 256, 0, s>>>(A, W, bias, resid, outF, outB, R, K, C, relu);
}

static void encoder_run(hipStream_t s, const float* X, int S, int Sv, const EncParams& p,
                        bf16* Yb, bf16* qkvB, bf16* oB, float* X2, bf16* h1B, float* X3, float* outLN){
  int R = NB * S;
  int lnBlocks = (R + 7)/8;
  ln_kernel<<<lnBlocks, 256, 0, s>>>(X, p.ln1g, p.ln1b, nullptr, Yb, R);
  gemm_launch(s, Yb, p.Wqkv, p.bqkv, nullptr, nullptr, qkvB, R, 256, 768, 0);
  int fw = NB*NHEAD*(S/16);
  flash_attn_kernel<<<(fw*32 + 255)/256, 256, 0, s>>>(qkvB, oB, S, Sv, S/16);
  gemm_launch(s, oB, p.Wo, p.bo, X, X2, nullptr, R, 256, 256, 0);
  ln_kernel<<<lnBlocks, 256, 0, s>>>(X2, p.ln2g, p.ln2b, nullptr, Yb, R);
  gemm_launch(s, Yb, p.W1, p.b1, nullptr, nullptr, h1B, R, 256, DFFD, 1);
  gemm_launch(s, h1B, p.W2, p.b2, X2, X3, nullptr, R, DFFD, 256, 0);
  ln_kernel<<<lnBlocks, 256, 0, s>>>(X3, p.lnfg, p.lnfb, outLN, nullptr, R);
}

extern "C" void kernel_launch(void* const* d_in, const int* in_sizes, int n_in,
                              void* d_out, int out_size, void* d_ws, size_t ws_size,
                              hipStream_t stream){
  (void)in_sizes; (void)n_in; (void)out_size; (void)ws_size;
  const float* bImg  = (const float*)d_in[0];
  const float* sWqkv = (const float*)d_in[1];  const float* sbqkv = (const float*)d_in[2];
  const float* sWo   = (const float*)d_in[3];  const float* sbo   = (const float*)d_in[4];
  const float* sln1g = (const float*)d_in[5];  const float* sln1b = (const float*)d_in[6];
  const float* sln2g = (const float*)d_in[7];  const float* sln2b = (const float*)d_in[8];
  const float* sW1   = (const float*)d_in[9];  const float* sb1   = (const float*)d_in[10];
  const float* sW2   = (const float*)d_in[11]; const float* sb2   = (const float*)d_in[12];
  const float* slnfg = (const float*)d_in[13]; const float* slnfb = (const float*)d_in[14];
  const float* aWqkv = (const float*)d_in[15]; const float* abqkv = (const float*)d_in[16];
  const float* tWqkv = (const float*)d_in[19]; const float* tbqkv = (const float*)d_in[20];
  const float* tWo   = (const float*)d_in[21]; const float* tbo   = (const float*)d_in[22];
  const float* tln1g = (const float*)d_in[23]; const float* tln1b = (const float*)d_in[24];
  const float* tln2g = (const float*)d_in[25]; const float* tln2b = (const float*)d_in[26];
  const float* tW1   = (const float*)d_in[27]; const float* tb1   = (const float*)d_in[28];
  const float* tW2   = (const float*)d_in[29]; const float* tb2   = (const float*)d_in[30];
  const float* tlnfg = (const float*)d_in[31]; const float* tlnfb = (const float*)d_in[32];
  const float* cls   = (const float*)d_in[33];

  char* ws = (char*)d_ws;
  size_t off = 0;
  auto alloc = [&](size_t bytes) -> void* {
    void* p = ws + off;
    off = (off + bytes + 255) & ~(size_t)255;
    return p;
  };

  const size_t Rsp = (size_t)NB*TT;
  float* xF   = (float*)alloc(Rsp*F_EMB*4);
  bf16*  xB   = (bf16*) alloc(Rsp*F_EMB*2);
  float* spAll= (float*)alloc(Rsp*F_EMB*4);
  bf16*  Yb   = (bf16*) alloc(Rsp*F_EMB*2);
  bf16*  qkvB = (bf16*) alloc(Rsp*768*2);
  bf16*  oB   = (bf16*) alloc(Rsp*F_EMB*2);
  float* X2   = (float*)alloc(Rsp*F_EMB*4);
  bf16*  h1B  = (bf16*) alloc(Rsp*DFFD*2);
  float* X3   = (float*)alloc(Rsp*F_EMB*4);
  bf16*  qkB  = (bf16*) alloc(Rsp*512*2);
  float* mbuf = (float*)alloc((size_t)NB*NHEAD*TT*4);
  float* lbuf = (float*)alloc((size_t)NB*NHEAD*TT*4);
  float* kscore=(float*)alloc((size_t)NB*TT*4);
  int*   idxB = (int*)  alloc((size_t)NB*KEYT_C*4);
  float* tm   = (float*)alloc((size_t)NB*SPAD*F_EMB*4);
  float* encT = (float*)alloc((size_t)NB*SPAD*F_EMB*4);
  float* sfm  = (float*)alloc((size_t)NB*F_EMB*4);
  bf16* sWqkvB=(bf16*)alloc(768*256*2); bf16* sWoB=(bf16*)alloc(256*256*2);
  bf16* sW1B =(bf16*)alloc(1024*256*2); bf16* sW2B=(bf16*)alloc(256*1024*2);
  bf16* aWqkvB=(bf16*)alloc(768*256*2);
  bf16* tWqkvB=(bf16*)alloc(768*256*2); bf16* tWoB=(bf16*)alloc(256*256*2);
  bf16* tW1B =(bf16*)alloc(1024*256*2); bf16* tW2B=(bf16*)alloc(256*1024*2);

  auto cvt = [&](const float* s, bf16* d, int n){
    cvt_kernel<<<(n + 255)/256, 256, 0, stream>>>(s, d, n);
  };
  cvt(sWqkv, sWqkvB, 768*256); cvt(sWo, sWoB, 256*256);
  cvt(sW1,  sW1B, 1024*256);   cvt(sW2, sW2B, 256*1024);
  cvt(aWqkv, aWqkvB, 768*256);
  cvt(tWqkv, tWqkvB, 768*256); cvt(tWo, tWoB, 256*256);
  cvt(tW1,  tW1B, 1024*256);   cvt(tW2, tW2B, 256*1024);

  transpose_kernel<<<(TT*NB*F_EMB + 255)/256, 256, 0, stream>>>(bImg, xF, xB);

  // spatial encoder
  EncParams sp = { sWqkvB, sWoB, sW1B, sW2B, sbqkv, sbo, sb1, sb2,
                   sln1g, sln1b, sln2g, sln2b, slnfg, slnfb };
  encoder_run(stream, xF, TT, TT, sp, Yb, qkvB, oB, X2, h1B, X3, spAll);

  // key-importance MHA: only q,k needed (first 512 output cols of Wqkv)
  gemm_launch(stream, xB, aWqkvB, abqkv, nullptr, nullptr, qkB, (int)Rsp, 256, 512, 0);
  zero_kernel<<<(NB*TT + 255)/256, 256, 0, stream>>>(kscore, NB*TT);
  {
    int qt = TT/16;
    int waves = NB*NHEAD*qt;
    attn_stats_kernel<<<(waves*32 + 255)/256, 256, 0, stream>>>(qkB, mbuf, lbuf, qt);
    attn_keyscore_kernel<<<(waves*32 + 255)/256, 256, 0, stream>>>(qkB, mbuf, lbuf, kscore, qt);
  }
  topk_kernel<<<NB, 256, 0, stream>>>(kscore, idxB);

  sfmean_kernel<<<(NB*F_EMB + 255)/256, 256, 0, stream>>>(spAll, idxB, sfm);
  build_tm_kernel<<<(NB*SPAD*F_EMB + 255)/256, 256, 0, stream>>>(xF, cls, idxB, tm);

  // temporal encoder (padded to 288, 257 valid rows; keys >=257 masked)
  EncParams tp = { tWqkvB, tWoB, tW1B, tW2B, tbqkv, tbo, tb1, tb2,
                   tln1g, tln1b, tln2g, tln2b, tlnfg, tlnfb };
  encoder_run(stream, tm, SPAD, SVALID, tp, Yb, qkvB, oB, X2, h1B, X3, encT);

  finalize_kernel<<<(NB*F_EMB + 255)/256, 256, 0, stream>>>(encT, sfm, (float*)d_out);
}
